// MHCA_43722767073910
// MI455X (gfx1250) — matrix operand order repaired and checked by reading, not compiled, not run on hardware
//
#include <hip/hip_runtime.h>
#include <math.h>
#include <stdint.h>

typedef _Float16 f16;
typedef __attribute__((ext_vector_type(16))) _Float16 v16h;
typedef __attribute__((ext_vector_type(8)))  _Float16 v8h;
typedef __attribute__((ext_vector_type(8)))  float    v8f;

#define S48 110592   // 48^3
#define S24 13824    // 24^3
#define S12 1728     // 12^3
#define NB  2
#define E   768
#define E4  192
#define CIN 6
#define L   1728
#define NH  12
#define HD  64
#define M1  3456     // NB * S12
#define EPSV 1e-6f

#define WMMA_F16(a, b, c) \
  __builtin_amdgcn_wmma_f32_16x16x32_f16(false, (a), false, (b), (short)0, (c), false, false)

__device__ __forceinline__ float gelu_exact(float x) {
  return 0.5f * x * (1.0f + erff(x * 0.70710678118654752f));
}

// ---- LDS byte offset of a shared-memory pointer (generic -> AS3 -> int) ----
__device__ __forceinline__ unsigned lds_addr(const void* p) {
  return (unsigned)(unsigned long long)(const __attribute__((address_space(3))) char*)p;
}

// ---- CDNA5 async global->LDS copy (ASYNCcnt-tracked; bypasses VGPRs) ----
__device__ __forceinline__ void async_b128(unsigned lds_off, const void* g) {
  asm volatile("global_load_async_to_lds_b128 %0, %1, off"
               :: "v"(lds_off), "v"((unsigned long long)(uintptr_t)g)
               : "memory");
}
__device__ __forceinline__ void wait_async0() {
  asm volatile("s_wait_asynccnt 0x0" ::: "memory");
}

// ---------------- WMMA fragment loads from LDS (contiguous 16B ds reads) ----------------
// A-style 16x32 fragment: lane row fixed; v0..3 = K[khalf*8 .. +8), v4..7 = K[16+khalf*8 .. +8)
__device__ __forceinline__ v16h lds_frag_a(const f16* rowp, int khalf) {
  v8h lo = *(const v8h*)(rowp + khalf * 8);
  v8h hi = *(const v8h*)(rowp + khalf * 8 + 16);
  v16h r;
#pragma unroll
  for (int j = 0; j < 8; ++j) { r[j] = lo[j]; r[j + 8] = hi[j]; }
  return r;
}
// B-style 32x16 fragment from K-contiguous (transposed) storage:
// lane col fixed; v0..3 = K[khalf*8 .. +8), v4..7 = K[16+khalf*8 .. +8)
__device__ __forceinline__ v16h lds_frag_b(const f16* rowp, int khalf) {
  v8h lo = *(const v8h*)(rowp + khalf * 8);
  v8h hi = *(const v8h*)(rowp + khalf * 8 + 16);
  v16h r;
#pragma unroll
  for (int j = 0; j < 8; ++j) { r[j] = lo[j]; r[j + 8] = hi[j]; }
  return r;
}

// ---------------- LayerNorm over 6 channels (channels_first) ----------------
__global__ void ln6_kernel(const float* __restrict__ in, const float* __restrict__ w,
                           const float* __restrict__ b, float* __restrict__ out) {
  long t = (long)blockIdx.x * 256 + threadIdx.x;
  if (t >= (long)NB * S48) return;
  int b_i = (int)(t / S48);
  int sp  = (int)(t % S48);
  const float* p = in + (long)b_i * CIN * S48 + sp;
  float v[CIN];
  float s = 0.f;
#pragma unroll
  for (int c = 0; c < CIN; ++c) { v[c] = p[(long)c * S48]; s += v[c]; }
  float mean = s * (1.f / CIN);
  float sq = 0.f;
#pragma unroll
  for (int c = 0; c < CIN; ++c) { float d = v[c] - mean; sq += d * d; }
  float rstd = rsqrtf(sq * (1.f / CIN) + EPSV);
  float* q = out + (long)b_i * CIN * S48 + sp;
#pragma unroll
  for (int c = 0; c < CIN; ++c) q[(long)c * S48] = (v[c] - mean) * rstd * w[c] + b[c];
}

// ---------------- conv1 (6->192, k2s2) + LN(192) + GELU, writes im2col A f16 ----------------
__global__ void conv1_ln_gelu(const float* __restrict__ xn, const float* __restrict__ w1,
                              const float* __restrict__ g1, const float* __restrict__ b1,
                              f16* __restrict__ A1) {
  __shared__ float sin48[48];
  __shared__ float red1[192], red2[192];
  int m24 = blockIdx.x;
  int tid = threadIdx.x;
  int b_i = m24 / S24;
  int sp  = m24 % S24;
  int y = sp / 576, x = (sp / 24) % 24, z = sp % 24;
  if (tid < 48) {
    int ic = tid >> 3, off = tid & 7;
    int kk = (off >> 2) & 1, pp = (off >> 1) & 1, qq = off & 1;
    sin48[tid] = xn[(long)b_i * CIN * S48 + (long)ic * S48 +
                    (2 * y + kk) * 2304 + (2 * x + pp) * 48 + (2 * z + qq)];
  }
  __syncthreads();
  float acc = 0.f;
#pragma unroll
  for (int j = 0; j < 48; ++j) acc += sin48[j] * w1[tid * 48 + j];
  red1[tid] = acc; red2[tid] = acc * acc;
  __syncthreads();
  if (tid < 64) {
    red1[tid] += red1[tid + 64] + red1[tid + 128];
    red2[tid] += red2[tid + 64] + red2[tid + 128];
  }
  __syncthreads();
  for (int st = 32; st > 0; st >>= 1) {
    if (tid < st) { red1[tid] += red1[tid + st]; red2[tid] += red2[tid + st]; }
    __syncthreads();
  }
  float mean = red1[0] * (1.f / 192.f);
  float var  = red2[0] * (1.f / 192.f) - mean * mean;
  float rstd = rsqrtf(var + EPSV);
  float h = gelu_exact((acc - mean) * rstd * g1[tid] + b1[tid]);
  long  m = (long)b_i * S12 + (y >> 1) * 144 + (x >> 1) * 12 + (z >> 1);
  int   off_p = ((y & 1) << 2) | ((x & 1) << 1) | (z & 1);
  A1[m * 1536 + tid * 8 + off_p] = (f16)h;
}

// ---------------- weight pack kernels ----------------
__global__ void pack_w2_kernel(const float* __restrict__ w, f16* __restrict__ o, int K, int N) {
  long t = (long)blockIdx.x * 256 + threadIdx.x;
  if (t >= (long)K * N) return;
  int k = (int)(t / N), n = (int)(t % N);
  o[t] = (f16)w[(long)n * K + k];
}
__global__ void cast_f16_kernel(const float* __restrict__ w, f16* __restrict__ o, long n) {
  long t = (long)blockIdx.x * 256 + threadIdx.x;
  if (t < n) o[t] = (f16)w[t];
}

// ---------------- WMMA GEMM with LDS-staged tiles ----------------
// block 128 (4 waves), block tile 64x64, K-step 32
// A staged row-major via async-to-LDS; B staged transposed [n][k] via b128 + scatter
__global__ void gemm_f16_wmma(const f16* __restrict__ A, const f16* __restrict__ B,
                              float* __restrict__ C, int M, int N, int K) {
  __shared__ alignas(16) f16 As[64][32];
  __shared__ alignas(16) f16 Bs[64][32];   // [n][k]
  int tid  = threadIdx.x;
  int lane = tid & 31, wave = tid >> 5;
  int mblk = blockIdx.y * 64;
  int nblk = blockIdx.x * 64;
  int r15 = lane & 15, khalf = lane >> 4;
  v8f acc0 = {}, acc1 = {}, acc2 = {}, acc3 = {};
  for (int k0 = 0; k0 < K; k0 += 32) {
    __syncthreads();   // protect previous tile reads
    {  // A: 32x32 halves = 2KB (rows 0..31 only) -> 128 threads x 16B, async direct copy
      int row = tid >> 2;
      int c8  = (tid & 3) * 8;
      async_b128(lds_addr(&As[row][c8]), A + (long)(mblk + row) * K + k0 + c8);
    }
#pragma unroll
    for (int c = 0; c < 2; ++c) {  // B: load b128 along N, scatter-transpose into Bs
      int chunk = tid * 2 + c;     // 0..255
      int k  = chunk >> 3;         // 0..31
      int n8 = (chunk & 7) * 8;    // 0..56
      v8h vb = *(const v8h*)(B + (long)(k0 + k) * N + nblk + n8);
#pragma unroll
      for (int j = 0; j < 8; ++j) Bs[n8 + j][k] = vb[j];
    }
    if (k0 + 32 < K) {             // prefetch next tiles (global_prefetch_b8)
      __builtin_prefetch(A + (long)(mblk + (tid >> 1)) * K + k0 + 32, 0, 1);
      __builtin_prefetch(B + (long)(k0 + 32 + lane) * N + nblk, 0, 1);
    }
    wait_async0();
    __syncthreads();
    v16h a  = lds_frag_a(&As[wave * 16 + r15][0], khalf);
    v16h b0 = lds_frag_b(&Bs[r15][0],      khalf);
    v16h b1 = lds_frag_b(&Bs[16 + r15][0], khalf);
    v16h b2 = lds_frag_b(&Bs[32 + r15][0], khalf);
    v16h b3 = lds_frag_b(&Bs[48 + r15][0], khalf);
    acc0 = WMMA_F16(a, b0, acc0);
    acc1 = WMMA_F16(a, b1, acc1);
    acc2 = WMMA_F16(a, b2, acc2);
    acc3 = WMMA_F16(a, b3, acc3);
  }
  int m0 = mblk + wave * 16;
  int rb = khalf * 8;
#pragma unroll
  for (int r = 0; r < 8; ++r) {
    long row = m0 + rb + r;
    C[row * N + nblk + r15]       = acc0[r];
    C[row * N + nblk + 16 + r15]  = acc1[r];
    C[row * N + nblk + 32 + r15]  = acc2[r];
    C[row * N + nblk + 48 + r15]  = acc3[r];
  }
}

// ---------------- row LN(768) + optional RoPE + scatter to head layout ----------------
// mode 0=Q (scale 1/8 + rope), 1=K (rope), 2=V  -> all [bh, L, 64] f16; mode 3 = plain f16 [m,768]
__global__ void ln_qkv_kernel(const float* __restrict__ Cmat, const float* __restrict__ g,
                              const float* __restrict__ bb, int mode, f16* __restrict__ out) {
  __shared__ float row[E];
  __shared__ float red1[256], red2[256];
  int m = blockIdx.x;
  int tid = threadIdx.x;
  int b_i = m / L, l = m % L;
  const float* cr = Cmat + (long)m * E;
  float s = 0.f, sq = 0.f;
  for (int c = tid; c < E; c += 256) { float v = cr[c]; row[c] = v; s += v; sq += v * v; }
  red1[tid] = s; red2[tid] = sq;
  __syncthreads();
  for (int st = 128; st > 0; st >>= 1) {
    if (tid < st) { red1[tid] += red1[tid + st]; red2[tid] += red2[tid + st]; }
    __syncthreads();
  }
  float mean = red1[0] * (1.f / E);
  float var  = red2[0] * (1.f / E) - mean * mean;
  float rstd = rsqrtf(var + EPSV);
  for (int c = tid; c < E; c += 256) row[c] = (row[c] - mean) * rstd * g[c] + bb[c];
  __syncthreads();
  for (int c = tid; c < E; c += 256) {
    float val;
    if (mode <= 1) {
      int j = (c < 384) ? c : c - 384;
      float inv = expf(-((float)(2 * j) / 768.f) * 9.210340371976184f);  // 10000^-(2j/768)
      float f = (float)l * inv;
      float co = cosf(f), si = sinf(f);
      float rot = (c < 384) ? -row[c + 384] : row[c - 384];
      val = row[c] * co + rot * si;
    } else {
      val = row[c];
    }
    if (mode <= 2) {
      if (mode == 0) val *= 0.125f;           // 1/sqrt(64) folded into Q
      int h = c / HD, d = c % HD;
      out[(((long)b_i * NH + h) * L + l) * HD + d] = (f16)val;
    } else {
      out[(long)m * E + c] = (f16)val;
    }
  }
}

// ---------------- flash attention with WMMA + LDS staging ----------------
// grid (L/64, NB*NH), block 128; each wave owns 16 query rows
__global__ void attn_wmma(const f16* __restrict__ QH, const f16* __restrict__ KH,
                          const f16* __restrict__ VH, float* __restrict__ OUT) {
  __shared__ alignas(16) f16 Qs[64][64];       // 8KB
  __shared__ alignas(16) f16 Ks[32][64];       // 4KB [key][d]
  __shared__ alignas(16) f16 Vt[64][32];       // 4KB [d][key]
  __shared__ alignas(16) f16 pbuf[4][16][32];  // 4KB
  int tid  = threadIdx.x;
  int lane = tid & 31, wave = tid >> 5;
  int bh = blockIdx.y, head = bh % NH, b_i = bh / NH;
  int qblk = blockIdx.x * 64;
  const f16* Qb = QH + (long)bh * L * HD;
  const f16* Kb = KH + (long)bh * L * HD;
  const f16* Vb = VH + (long)bh * L * HD;
  int r15 = lane & 15, khalf = lane >> 4;
  // stage Q tile once (async direct copy, 8KB)
#pragma unroll
  for (int c = 0; c < 4; ++c) {
    int chunk = tid * 4 + c;     // 0..511
    int qr = chunk >> 3, d8 = (chunk & 7) * 8;
    async_b128(lds_addr(&Qs[qr][d8]), Qb + (long)(qblk + qr) * HD + d8);
  }
  wait_async0();
  __syncthreads();
  v16h aq0 = lds_frag_a(&Qs[wave * 16 + r15][0],  khalf);
  v16h aq1 = lds_frag_a(&Qs[wave * 16 + r15][32], khalf);
  float mrow[8], lrow[8];
#pragma unroll
  for (int r = 0; r < 8; ++r) { mrow[r] = -INFINITY; lrow[r] = 0.f; }
  v8f o0 = {}, o1 = {}, o2 = {}, o3 = {};
  for (int kt = 0; kt < L; kt += 32) {
    __syncthreads();   // protect previous K/V tiles
#pragma unroll
    for (int c = 0; c < 2; ++c) {  // K tile: async direct copy (row-major)
      int chunk = tid * 2 + c;
      int key = chunk >> 3, d8 = (chunk & 7) * 8;
      async_b128(lds_addr(&Ks[key][d8]), Kb + (long)(kt + key) * HD + d8);
    }
#pragma unroll
    for (int c = 0; c < 2; ++c) {  // V tile: b128 load + scatter transpose
      int chunk = tid * 2 + c;
      int key = chunk >> 3, d8 = (chunk & 7) * 8;
      v8h vv = *(const v8h*)(Vb + (long)(kt + key) * HD + d8);
#pragma unroll
      for (int j = 0; j < 8; ++j) Vt[d8 + j][key] = vv[j];
    }
    wait_async0();
    __syncthreads();
    // S tile 16x32 (2 column halves x 2 k-steps over hd=64)
    v8f s0 = {}, s1 = {};
    v16h bk;
    bk = lds_frag_b(&Ks[r15][0],       khalf); s0 = WMMA_F16(aq0, bk, s0);
    bk = lds_frag_b(&Ks[r15][32],      khalf); s0 = WMMA_F16(aq1, bk, s0);
    bk = lds_frag_b(&Ks[16 + r15][0],  khalf); s1 = WMMA_F16(aq0, bk, s1);
    bk = lds_frag_b(&Ks[16 + r15][32], khalf); s1 = WMMA_F16(aq1, bk, s1);
    // online softmax; lane half owns rows rb..rb+7 at its column
    int rb = khalf * 8;
    float alpha[8];
#pragma unroll
    for (int r = 0; r < 8; ++r) {
      float tm = fmaxf(s0[r], s1[r]);
#pragma unroll
      for (int mk = 1; mk < 16; mk <<= 1) tm = fmaxf(tm, __shfl_xor(tm, mk, 32));
      float mnew = fmaxf(mrow[r], tm);
      float p0 = expf(s0[r] - mnew);
      float p1 = expf(s1[r] - mnew);
      float rs = p0 + p1;
#pragma unroll
      for (int mk = 1; mk < 16; mk <<= 1) rs += __shfl_xor(rs, mk, 32);
      float al = expf(mrow[r] - mnew);
      lrow[r] = lrow[r] * al + rs;
      mrow[r] = mnew;
      alpha[r] = al;
      pbuf[wave][rb + r][r15]      = (f16)p0;
      pbuf[wave][rb + r][r15 + 16] = (f16)p1;
    }
#pragma unroll
    for (int r = 0; r < 8; ++r) {
      o0[r] *= alpha[r]; o1[r] *= alpha[r]; o2[r] *= alpha[r]; o3[r] *= alpha[r];
    }
    v16h ap = lds_frag_a(&pbuf[wave][r15][0], khalf);
    v16h bv;
    bv = lds_frag_b(&Vt[r15][0],      khalf); o0 = WMMA_F16(ap, bv, o0);
    bv = lds_frag_b(&Vt[16 + r15][0], khalf); o1 = WMMA_F16(ap, bv, o1);
    bv = lds_frag_b(&Vt[32 + r15][0], khalf); o2 = WMMA_F16(ap, bv, o2);
    bv = lds_frag_b(&Vt[48 + r15][0], khalf); o3 = WMMA_F16(ap, bv, o3);
  }
  // epilogue: divide by l, scatter with channel = d*NH + head  ('b nh l c -> b (c nh) ...')
  int rb = khalf * 8;
#pragma unroll
  for (int r = 0; r < 8; ++r) {
    int lq = qblk + wave * 16 + rb + r;
    long base = ((long)b_i * L + lq) * E;
    float inv_l = 1.f / lrow[r];
    OUT[base + (0 * 16 + r15) * NH + head] = o0[r] * inv_l;
    OUT[base + (1 * 16 + r15) * NH + head] = o1[r] * inv_l;
    OUT[base + (2 * 16 + r15) * NH + head] = o2[r] * inv_l;
    OUT[base + (3 * 16 + r15) * NH + head] = o3[r] * inv_l;
  }
}

// ---------------- final: LN(192)+GELU on convT1 output, convT2 (192->6) + bias + residual ----------------
__global__ void debed_final(const float* __restrict__ C2, const float* __restrict__ og,
                            const float* __restrict__ ob, const float* __restrict__ tw2,
                            const float* __restrict__ tb2, const float* __restrict__ resid,
                            float* __restrict__ out) {
  __shared__ float sh[192];
  __shared__ float red1[192], red2[192];
  int m24 = blockIdx.x;
  int tid = threadIdx.x;
  int b_i = m24 / S24;
  int sp  = m24 % S24;
  int y = sp / 576, x = (sp / 24) % 24, z = sp % 24;
  long m12  = (long)b_i * S12 + (y >> 1) * 144 + (x >> 1) * 12 + (z >> 1);
  int  offp = ((y & 1) << 2) | ((x & 1) << 1) | (z & 1);
  float v = C2[m12 * 1536 + tid * 8 + offp];
  red1[tid] = v; red2[tid] = v * v;
  __syncthreads();
  if (tid < 64) {
    red1[tid] += red1[tid + 64] + red1[tid + 128];
    red2[tid] += red2[tid + 64] + red2[tid + 128];
  }
  __syncthreads();
  for (int st = 32; st > 0; st >>= 1) {
    if (tid < st) { red1[tid] += red1[tid + st]; red2[tid] += red2[tid + st]; }
    __syncthreads();
  }
  float mean = red1[0] * (1.f / 192.f);
  float var  = red2[0] * (1.f / 192.f) - mean * mean;
  float rstd = rsqrtf(var + EPSV);
  sh[tid] = gelu_exact((v - mean) * rstd * og[tid] + ob[tid]);
  __syncthreads();
  if (tid < 48) {
    int oc = tid >> 3, off2 = tid & 7;
    int kk = (off2 >> 2) & 1, pp = (off2 >> 1) & 1, qq = off2 & 1;
    float acc = 0.f;
#pragma unroll 8
    for (int o = 0; o < 192; ++o) acc += sh[o] * tw2[o * 48 + tid];
    int Y = 2 * y + kk, X = 2 * x + pp, Z = 2 * z + qq;
    long idx = (long)b_i * CIN * S48 + (long)oc * S48 + Y * 2304 + X * 48 + Z;
    out[idx] = acc + tb2[oc] + resid[idx];
  }
}

// ---------------- host launcher ----------------
extern "C" void kernel_launch(void* const* d_in, const int* in_sizes, int n_in,
                              void* d_out, int out_size, void* d_ws, size_t ws_size,
                              hipStream_t stream) {
  (void)in_sizes; (void)n_in; (void)out_size; (void)ws_size;
  char* ws = (char*)d_ws;
  const size_t SZ_XN  = (size_t)NB * CIN * S48 * 4;     // 5,308,416
  const size_t SZ_A1  = (size_t)M1 * 1536 * 2;          // 10,616,832
  const size_t SZ_B1  = (size_t)1536 * 768 * 2;         // 2,359,296
  const size_t SZ_C1  = (size_t)M1 * 768 * 4;           // 10,616,832
  const size_t SZ_QH  = (size_t)NB * NH * L * HD * 2;   // 5,308,416

  float* xn  = (float*)(ws);
  float* skn = (float*)(ws + SZ_XN);
  f16*   A1  = (f16*)  (ws + 2 * SZ_XN);
  f16*   B1  = (f16*)  (ws + 2 * SZ_XN + SZ_A1);
  float* C1  = (float*)(ws + 2 * SZ_XN + SZ_A1 + SZ_B1);            // also attn output
  f16*   QH  = (f16*)  (ws + 2 * SZ_XN + SZ_A1 + SZ_B1 + SZ_C1);
  f16*   KH  = (f16*)  ((char*)QH + SZ_QH);
  f16*   VH  = (f16*)  ((char*)KH + SZ_QH);
  float* C2  = (float*)((char*)VH + SZ_QH);

  const float* x_in    = (const float*)d_in[0];
  const float* skip_in = (const float*)d_in[1];

  // 1) channel LN on x and skip
  int gln = (NB * S48 + 255) / 256;
  ln6_kernel<<<gln, 256, 0, stream>>>(skip_in, (const float*)d_in[25], (const float*)d_in[26], skn);
  ln6_kernel<<<gln, 256, 0, stream>>>(x_in,    (const float*)d_in[27], (const float*)d_in[28], xn);

  // 2) three stems: conv1+LN+GELU -> WMMA GEMM -> LN+rope+scatter
  const float* stem_in[3] = { skn, xn, xn };           // q from skip, k/v from x
  const int    wbase[3]   = { 2, 8, 14 };
  f16* qkv_out[3] = { QH, KH, VH };
  for (int s = 0; s < 3; ++s) {
    conv1_ln_gelu<<<NB * S24, 192, 0, stream>>>(
        stem_in[s], (const float*)d_in[wbase[s]],
        (const float*)d_in[wbase[s] + 1], (const float*)d_in[wbase[s] + 2], A1);
    pack_w2_kernel<<<(1536 * 768 + 255) / 256, 256, 0, stream>>>(
        (const float*)d_in[wbase[s] + 3], B1, 1536, 768);
    gemm_f16_wmma<<<dim3(768 / 64, M1 / 64), 128, 0, stream>>>(A1, B1, C1, M1, 768, 1536);
    ln_qkv_kernel<<<M1, 256, 0, stream>>>(
        C1, (const float*)d_in[wbase[s] + 4], (const float*)d_in[wbase[s] + 5], s, qkv_out[s]);
  }

  // 3) attention (flash style, WMMA), writes [b, l, 768] with channel = d*12 + head
  attn_wmma<<<dim3(L / 64, NB * NH), 128, 0, stream>>>(QH, KH, VH, C1);

  // 4) LN(768) -> f16 A2 (reuse A1 region)
  ln_qkv_kernel<<<M1, 256, 0, stream>>>(C1, (const float*)d_in[29], (const float*)d_in[30], 3, A1);

  // 5) pack o_tw1 (identity layout cast) -> B2 [768,1536] (reuse B1 region)
  cast_f16_kernel<<<(768 * 1536 + 255) / 256, 256, 0, stream>>>((const float*)d_in[20], B1,
                                                                (long)768 * 1536);

  // 6) debed GEMM: [3456,1536] = [3456,768] x [768,1536]
  gemm_f16_wmma<<<dim3(1536 / 64, M1 / 64), 128, 0, stream>>>(A1, B1, C2, M1, 1536, 768);

  // 7) LN(192)+GELU + convT2 + bias + residual
  debed_final<<<NB * S24, 192, 0, stream>>>(
      C2, (const float*)d_in[21], (const float*)d_in[22],
      (const float*)d_in[23], (const float*)d_in[24], skip_in, (float*)d_out);
}
